// MultiheadAttentionMechanism_28097676050753
// MI455X (gfx1250) — matrix operand order repaired and checked by reading, not compiled, not run on hardware
//
#include <hip/hip_runtime.h>

// dims
#define B_ 16
#define T_ 1500
#define E_ 512
#define D_ 512
#define A_ 512
#define C_ 10
#define K_ 201
#define H_ 4

// A-tile staging: 544 useful K-halves (512 enc + 32 conv), padded to 552 so the
// row stride is 276 dwords -> 16 lanes' ds_load_b128 hit all 64 LDS banks once.
#define AK_   544
#define AKP_  552

typedef __attribute__((ext_vector_type(16))) _Float16 v16h;
typedef __attribute__((ext_vector_type(8)))  _Float16 v8h;
typedef __attribute__((ext_vector_type(4)))  _Float16 v4h;
typedef __attribute__((ext_vector_type(8)))  float    v8f;
typedef __attribute__((ext_vector_type(4)))  int      v4i;

#if __has_builtin(__builtin_amdgcn_global_load_async_to_lds_b128)
#define HAVE_ASYNC_LDS 1
typedef __attribute__((address_space(1))) v4i* g4_t;   // global int4*
typedef __attribute__((address_space(3))) v4i* l4_t;   // LDS int4*
#endif

// load a 16-half WMMA fragment as two b128 loads separated by `step` halves
__device__ __forceinline__ v16h load_frag(const _Float16* p, int step) {
    v8h lo = *(const v8h*)p;
    v8h hi = *(const v8h*)(p + step);
    v16h f;
#pragma unroll
    for (int i = 0; i < 8; ++i) { f[i] = lo[i]; f[8 + i] = hi[i]; }
    return f;
}

// ---------------- prep kernels ----------------

// f32 -> f16 convert of enc_out, 4 elements/thread
__global__ void k_cvt_enc(const float* __restrict__ in, _Float16* __restrict__ out, int n4) {
    int i = blockIdx.x * 256 + threadIdx.x;
    if (i < n4) {
        float4 x = ((const float4*)in)[i];
        v4h y;
        y[0] = (_Float16)x.x; y[1] = (_Float16)x.y;
        y[2] = (_Float16)x.z; y[3] = (_Float16)x.w;
        ((v4h*)out)[i] = y;
    }
}

// W_enc [h][e][a] f32 -> W_encT [h][a][e] f16 (B-fragment friendly: contiguous K per column)
__global__ void k_wencT(const float* __restrict__ W_enc, _Float16* __restrict__ out) {
    int tid = blockIdx.x * 256 + threadIdx.x;          // tid == ((h*A + a)*E + e)
    if (tid < H_ * A_ * E_) {
        int e = tid & 511;
        int a = (tid >> 9) & 511;
        int h = tid >> 18;
        out[tid] = (_Float16)W_enc[((size_t)h * E_ + e) * A_ + a];
    }
}

// W_conv [h][c][a] f32 -> padded [h][a][32] f16 (c >= C zero) for augmented K-chunk
__global__ void k_wcpad(const float* __restrict__ W_conv, _Float16* __restrict__ out) {
    int tid = blockIdx.x * 256 + threadIdx.x;          // tid == ((h*A + a)*32 + cc)
    if (tid < H_ * A_ * 32) {
        int cc = tid & 31;
        int a  = (tid >> 5) & 511;
        int h  = tid >> 14;
        float val = (cc < C_) ? W_conv[((size_t)h * C_ + cc) * A_ + a] : 0.0f;
        out[tid] = (_Float16)val;
    }
}

// bias[b][h][a] = b_enc[h][a] + dec_out[b] . W_dec[h][:,a]
__global__ void k_bias(const float* __restrict__ dec_out, const float* __restrict__ W_dec,
                       const float* __restrict__ b_enc, float* __restrict__ bias) {
    int tid = blockIdx.x * 256 + threadIdx.x;          // n = B*H*A
    int a = tid & 511;
    int h = (tid >> 9) & 3;
    int b = tid >> 11;
    float acc = b_enc[h * A_ + a];
    const float* dec = dec_out + (size_t)b * D_;
    const float* w   = W_dec + (size_t)h * D_ * A_ + a;
    for (int d = 0; d < D_; ++d) acc += dec[d] * w[(size_t)d * A_];
    bias[tid] = acc;                                   // layout ((b*H+h)*A + a)
}

// location conv: conv_feat[b][h][t][0..31] f16, channels 10..31 zero
__global__ void k_convfeat(const float* __restrict__ aw_step, const float* __restrict__ conv_w,
                           _Float16* __restrict__ convf) {
    int tid = blockIdx.x * 256 + threadIdx.x;          // tid == ((b*H+h)*T + t)
    if (tid >= B_ * H_ * T_) return;
    int t  = tid % T_;
    int bh = tid / T_;
    int h  = bh % H_;
    int b  = bh / H_;
    float acc[C_];
#pragma unroll
    for (int c = 0; c < C_; ++c) acc[c] = 0.0f;
    for (int k = 0; k < K_; ++k) {
        int tt = t + k - (K_ / 2);                     // SAME padding, pad=100
        if (tt < 0 || tt >= T_) continue;
        float x = aw_step[((size_t)b * T_ + tt) * H_ + h];
#pragma unroll
        for (int c = 0; c < C_; ++c) acc[c] += x * conv_w[((size_t)h * C_ + c) * K_ + k];
    }
    _Float16* o = convf + (size_t)tid * 32;
    v8h o0, o1, z = {};
#pragma unroll
    for (int i = 0; i < 8; ++i) o0[i] = (_Float16)acc[i];
    o1 = z;
    o1[0] = (_Float16)acc[8];
    o1[1] = (_Float16)acc[9];
    *(v8h*)(o + 0)  = o0;
    *(v8h*)(o + 8)  = o1;
    *(v8h*)(o + 16) = z;
    *(v8h*)(o + 24) = z;
}

// ---------------- fused WMMA energy kernel ----------------
// One block per (t-tile16, h, b). 8 waves; wave w owns a-tiles w*4..w*4+3.
// A tile (16 rows x 544 K halves: enc 512 + conv-pad 32) is staged once per block
// into LDS via async-to-LDS; fragments come from ds_load_b128. The K loop is
// software-pipelined: A(kc+1)/B(kc+1) fragments are issued before the WMMAs of kc.
// energy[b,h,t] = mask * sum_a v[h,a] * tanh(GEMM + bias)
__global__ void __launch_bounds__(256) k_energy(
        const _Float16* __restrict__ enc_h,
        const _Float16* __restrict__ wencT,
        const _Float16* __restrict__ convf,
        const _Float16* __restrict__ wcpad,
        const float* __restrict__ bias,
        const float* __restrict__ v,
        const int* __restrict__ x_lens,
        float* __restrict__ energy) {
    const int t0   = blockIdx.x * 16;
    const int h    = blockIdx.y;
    const int b    = blockIdx.z;
    const int lane = threadIdx.x & 31;
    const int wave = threadIdx.x >> 5;
    const int ln16 = lane & 15;
    const bool hih = lane >= 16;

    __shared__ _Float16 As[16][AKP_];   // 17,664 B
    __shared__ float ered[16];

    // ---- cooperative A-tile staging: 16 rows x 68 chunks of 8 halves ----
    for (int i = 0; i < 5; ++i) {
        int cidx = threadIdx.x + 256 * i;              // 0..1087
        if (cidx < 16 * 68) {
            int row = cidx / 68;
            int c   = cidx - row * 68;
            int off = c * 8;
            int tr  = t0 + row;
            if (tr >= T_) tr = T_ - 1;                 // clamp; masked on write
            const _Float16* g = (c < 64)
                ? (enc_h + ((size_t)b * T_ + tr) * E_ + off)
                : (convf + (((size_t)b * H_ + h) * T_ + tr) * 32 + (off - 512));
#if defined(HAVE_ASYNC_LDS)
            __builtin_amdgcn_global_load_async_to_lds_b128(
                (g4_t)g, (l4_t)&As[row][off], 0, 0);
#else
            *(v8h*)&As[row][off] = *(const v8h*)g;
#endif
        }
    }
#if defined(HAVE_ASYNC_LDS)
#if __has_builtin(__builtin_amdgcn_s_wait_asynccnt)
    __builtin_amdgcn_s_wait_asynccnt(0);
#else
    asm volatile("s_wait_asynccnt 0x0" ::: "memory");
#endif
#endif
    if (threadIdx.x < 16) ered[threadIdx.x] = 0.0f;
    __syncthreads();

    const int aoff = hih ? 16 : 0;                     // A frag: lanes>=16 hold K 16..31
    const int boff = hih ? 16 : 0;                     // B frag: lanes>=16 hold K 16..31
    const int col_base = wave * 64 + ln16;             // wave's first a-column for this lane

    // B pointers per a-tile j: wencT rows for kc<16, wcpad row for kc==16
    const _Float16* brow[4];
#pragma unroll
    for (int j = 0; j < 4; ++j)
        brow[j] = wencT + ((size_t)h * A_ + (col_base + j * 16)) * E_ + boff;
    const _Float16* crow[4];
#pragma unroll
    for (int j = 0; j < 4; ++j)
        crow[j] = wcpad + ((size_t)h * A_ + (col_base + j * 16)) * 32 + boff;

    v8f acc[4] = {};

    // prologue: fragments for chunk 0
    v16h acur = load_frag(&As[ln16][aoff], 8);
    v16h bcur[4];
#pragma unroll
    for (int j = 0; j < 4; ++j) bcur[j] = load_frag(brow[j], 8);

#pragma unroll
    for (int kc = 0; kc < 17; ++kc) {
        v16h anxt;
        v16h bnxt[4];
        if (kc + 1 < 17) {
            // issue next chunk's loads before this chunk's WMMAs
            anxt = load_frag(&As[ln16][(kc + 1) * 32 + aoff], 8);
#pragma unroll
            for (int j = 0; j < 4; ++j)
                bnxt[j] = (kc + 1 < 16) ? load_frag(brow[j] + (kc + 1) * 32, 8)
                                        : load_frag(crow[j], 8);
            if (kc + 2 < 16)
                __builtin_prefetch(brow[0] + (kc + 2) * 32, 0, 1);
        }
#pragma unroll
        for (int j = 0; j < 4; ++j)
            acc[j] = __builtin_amdgcn_wmma_f32_16x16x32_f16(
                false, acur, false, bcur[j], (short)0, acc[j], false, false);
        if (kc + 1 < 17) {
            acur = anxt;
#pragma unroll
            for (int j = 0; j < 4; ++j) bcur[j] = bnxt[j];
        }
    }

    // epilogue: + bias, tanh, * v, reduce over a (16 lanes per half + LDS atomics)
#pragma unroll
    for (int j = 0; j < 4; ++j) {
        const int a_col = col_base + j * 16;
        const float bb = bias[((size_t)b * H_ + h) * A_ + a_col];
        const float vv = v[h * A_ + a_col];
#pragma unroll
        for (int r = 0; r < 8; ++r) {
            float val = vv * tanhf(acc[j][r] + bb);
            val += __shfl_xor(val, 1);
            val += __shfl_xor(val, 2);
            val += __shfl_xor(val, 4);
            val += __shfl_xor(val, 8);
            if (lane == 0)       atomicAdd(&ered[r], val);      // row M = r
            else if (lane == 16) atomicAdd(&ered[r + 8], val);  // row M = r+8
        }
    }
    __syncthreads();
    if (threadIdx.x < 16) {
        int t = t0 + threadIdx.x;
        if (t < T_) {
            float m = (t < x_lens[b]) ? 1.0f : 0.0f;   // multiplicative mask, as in reference
            energy[((size_t)b * H_ + h) * T_ + t] = ered[threadIdx.x] * m;
        }
    }
}

// ---------------- softmax over T per (b,h); masked slots DO participate ----------------
__global__ void k_softmax(const float* __restrict__ energy, float* __restrict__ aw) {
    const int bh = blockIdx.x;
    const int b = bh / H_, h = bh % H_;
    const float* e = energy + (size_t)bh * T_;
    __shared__ float red[256];
    float mx = -1e30f;
    for (int t = threadIdx.x; t < T_; t += 256) mx = fmaxf(mx, e[t]);
    red[threadIdx.x] = mx; __syncthreads();
    for (int s = 128; s > 0; s >>= 1) {
        if (threadIdx.x < s) red[threadIdx.x] = fmaxf(red[threadIdx.x], red[threadIdx.x + s]);
        __syncthreads();
    }
    mx = red[0]; __syncthreads();
    float sm = 0.0f;
    for (int t = threadIdx.x; t < T_; t += 256) sm += expf(e[t] - mx);
    red[threadIdx.x] = sm; __syncthreads();
    for (int s = 128; s > 0; s >>= 1) {
        if (threadIdx.x < s) red[threadIdx.x] += red[threadIdx.x + s];
        __syncthreads();
    }
    float inv = 1.0f / red[0];
    for (int t = threadIdx.x; t < T_; t += 256)
        aw[((size_t)b * T_ + t) * H_ + h] = expf(e[t] - mx) * inv;
}

// ---------------- context: ctx[b][h][e] = sum_t aw[b,t,h] * enc[b,t,e] ----------------
__global__ void k_ctx_accum(const float* __restrict__ enc, const float* __restrict__ aw,
                            float* __restrict__ ctx) {
    const int b = blockIdx.y;
    const int e = blockIdx.x * 256 + threadIdx.x;
    float a0 = 0, a1 = 0, a2 = 0, a3 = 0;
    for (int t = 0; t < T_; ++t) {
        float x = enc[((size_t)b * T_ + t) * E_ + e];
        const float4 w = *(const float4*)(aw + ((size_t)b * T_ + t) * H_); // H=4 contiguous
        a0 += x * w.x; a1 += x * w.y; a2 += x * w.z; a3 += x * w.w;
    }
    ctx[((size_t)b * H_ + 0) * E_ + e] = a0;
    ctx[((size_t)b * H_ + 1) * E_ + e] = a1;
    ctx[((size_t)b * H_ + 2) * E_ + e] = a2;
    ctx[((size_t)b * H_ + 3) * E_ + e] = a3;
}

// ---------------- final: context[b][e] = ctx[b] . W_mha[:,e] + b_mha[e] ----------------
__global__ void k_ctx_out(const float* __restrict__ ctx, const float* __restrict__ W_mha,
                          const float* __restrict__ b_mha, float* __restrict__ out) {
    int tid = blockIdx.x * 256 + threadIdx.x;          // n = B*E
    int b = tid >> 9, e = tid & 511;
    float acc = b_mha[e];
    const float* c = ctx + (size_t)b * (H_ * E_);
    for (int k = 0; k < H_ * E_; ++k) acc += c[k] * W_mha[(size_t)k * E_ + e];
    out[(size_t)b * E_ + e] = acc;
}

extern "C" void kernel_launch(void* const* d_in, const int* in_sizes, int n_in,
                              void* d_out, int out_size, void* d_ws, size_t ws_size,
                              hipStream_t stream) {
    const float* enc_out = (const float*)d_in[0];
    const int*   x_lens  = (const int*)  d_in[1];
    const float* dec_out = (const float*)d_in[2];
    const float* aw_step = (const float*)d_in[3];
    const float* W_enc   = (const float*)d_in[4];
    const float* b_enc   = (const float*)d_in[5];
    const float* W_dec   = (const float*)d_in[6];
    const float* W_conv  = (const float*)d_in[7];
    const float* conv_w  = (const float*)d_in[8];
    const float* v       = (const float*)d_in[9];
    const float* W_mha   = (const float*)d_in[10];
    const float* b_mha   = (const float*)d_in[11];

    char* ws = (char*)d_ws;                         // all offsets 256B-aligned
    _Float16* enc_h  = (_Float16*)(ws + 0);         // 24,576,000 B
    _Float16* wencT  = (_Float16*)(ws + 24576000);  //  2,097,152 B
    _Float16* wcpad  = (_Float16*)(ws + 26673152);  //    131,072 B
    _Float16* convf  = (_Float16*)(ws + 26804224);  //  6,144,000 B
    float*    bias   = (float*)   (ws + 32948224);  //    131,072 B
    float*    energy = (float*)   (ws + 33079296);  //    384,000 B
    float*    ctx    = (float*)   (ws + 33463296);  //    131,072 B

    float* out_ctx = (float*)d_out;                 // [B,1,E]   = 8192 floats
    float* out_aw  = (float*)d_out + B_ * E_;       // [B,T,H]   = 96000 floats

    k_cvt_enc <<<12000, 256, 0, stream>>>(enc_out, enc_h, (B_ * T_ * E_) / 4);
    k_wencT   <<<4096,  256, 0, stream>>>(W_enc, wencT);
    k_wcpad   <<<256,   256, 0, stream>>>(W_conv, wcpad);
    k_bias    <<<128,   256, 0, stream>>>(dec_out, W_dec, b_enc, bias);
    k_convfeat<<<375,   256, 0, stream>>>(aw_step, conv_w, convf);
    k_energy  <<<dim3((T_ + 15) / 16, H_, B_), 256, 0, stream>>>(
        enc_h, wencT, convf, wcpad, bias, v, x_lens, energy);
    k_softmax <<<B_ * H_, 256, 0, stream>>>(energy, out_aw);
    k_ctx_accum<<<dim3(E_ / 256, B_), 256, 0, stream>>>(enc_out, out_aw, ctx);
    k_ctx_out <<<(B_ * E_) / 256, 256, 0, stream>>>(ctx, W_mha, b_mha, out_ctx);
}
